// MambaLayer_53824530154111
// MI455X (gfx1250) — matrix operand order repaired and checked by reading, not compiled, not run on hardware
//
#include <hip/hip_runtime.h>
#include <hip/hip_bf16.h>
#include <math.h>

// ---------------------------------------------------------------------------
// Problem constants (from reference): B=4, N=4096 (64x64), DM=DIN=256,
// DS=16 (state), DC=4 (conv taps), DTR=16 (dt rank).
// ---------------------------------------------------------------------------
#define B_   4
#define N_   4096
#define DM_  256
#define DIN_ 256
#define DS_  16
#define DC_  4
#define DTR_ 16
#define HH_  64
#define WW_  64
#define XPJ_ (DTR_ + 2 * DS_)   // 48
#define KK_  256                // K dim of both WMMA GEMMs (DM_ == DIN_)
#define KSTEPS_ (KK_ / 32)      // 8

typedef __attribute__((ext_vector_type(16))) __bf16 v16bf;
typedef __attribute__((ext_vector_type(8)))  float  v8f;
typedef int v4i_ __attribute__((vector_size(16)));   // matches builtin proto

// CDNA5 async global->LDS copy path (ASYNCcnt-tracked), guarded so the file
// still compiles on toolchains that lack the builtins.  The builtin takes
// (AS1 int4* gsrc, AS3 int4* ldst, imm offset, imm cpol).
#if __has_builtin(__builtin_amdgcn_global_load_async_to_lds_b128) && \
    __has_builtin(__builtin_amdgcn_s_wait_asynccnt)
#define USE_ASYNC_LDS 1
#define GPTR(p) ((__attribute__((address_space(1))) v4i_*)(p))
#define LPTR(p) ((__attribute__((address_space(3))) v4i_*)(p))
#else
#define USE_ASYNC_LDS 0
#endif

union FragU {                 // two 16B LDS vector loads -> one WMMA fragment
  v16bf v;
  uint4 q[2];
};

struct bh4 { __bf16 x, y, z, w; };   // 8-byte packed bf16 quad

// Token permutation for each scan direction (sequence pos -> token index).
__device__ __forceinline__ int perm_tok(int t, int dir) {
  switch (dir) {
    case 0:  return t;                                   // horizontal fwd
    case 1:  return N_ - 1 - t;                          // horizontal bwd
    case 2:  return (t & (HH_ - 1)) * WW_ + (t >> 6);    // vertical fwd
    default: { int tv = N_ - 1 - t;                      // vertical bwd
               return (tv & (HH_ - 1)) * WW_ + (tv >> 6); }
  }
}

// ---------------------------------------------------------------------------
// fp32 -> bf16 (with scale), vectorized 4-wide.
// ---------------------------------------------------------------------------
__global__ __launch_bounds__(256) void cvt_bf16_scale(
    const float* __restrict__ in, __bf16* __restrict__ out, float scale,
    int n4) {
  int i = blockIdx.x * 256 + threadIdx.x;
  if (i < n4) {
    float4 v = ((const float4*)in)[i];
    bh4 o;
    o.x = (__bf16)(v.x * scale); o.y = (__bf16)(v.y * scale);
    o.z = (__bf16)(v.z * scale); o.w = (__bf16)(v.w * scale);
    ((bh4*)out)[i] = o;
  }
}

// Weight transpose + convert: WT[n][k] = (bf16)W[k][n].  Tiny (<=0.5MB).
__global__ __launch_bounds__(256) void transpose_cvt_bf16(
    const float* __restrict__ W, __bf16* __restrict__ WT, int K, int Ncols) {
  int i = blockIdx.x * 256 + threadIdx.x;
  if (i < K * Ncols) {
    int n = i / K, k = i - n * K;
    WT[(size_t)n * K + k] = (__bf16)W[(size_t)k * Ncols + n];
  }
}

// ---------------------------------------------------------------------------
// bf16 WMMA GEMM:  C[M,N] = A[M,KK_] @ BT[N,KK_]^T  (bf16 in, fp32 out).
// A row-major bf16; BT is B pre-transposed ([n][k]) so both tiles stage as
// straight 16B copies.  Double-buffered LDS + async global->LDS DMA: tile
// kb+1's copies stay in flight (partial s_wait_asynccnt 2) while tile kb's
// WMMAs execute.  Every WMMA fragment is exactly two ds_load_b128 reads.
// Optional epilogue: nan_to_num + clip(+-1000) to match the reference tail.
// ---------------------------------------------------------------------------
__global__ __launch_bounds__(256) void gemm_bf16_wmma(
    const __bf16* __restrict__ A, const __bf16* __restrict__ BT,
    float* __restrict__ C, int N, int do_clip) {
  __shared__ alignas(16) __bf16 As [2][64][40];  // 80B pitch, 16B-aligned
  __shared__ alignas(16) __bf16 BsT[2][64][40];  // B tile, [n][k]

  const int tid   = threadIdx.x;
  const int lane  = tid & 31;
  const int wave  = tid >> 5;
  const int mBase = blockIdx.y * 64;
  const int nBase = blockIdx.x * 64;

  const int tr     = wave & 3;          // 16-row tile within the block
  const int tcPair = wave >> 2;         // which pair of 16-col tiles
  const int khalf  = (lane >> 4) << 3;  // A lane K offset (0 or 8)
  const int kboff  = (lane >> 4) << 3;  // B lane K offset (0 or 8)
  const int l15    = lane & 15;
  const int mrow   = tr * 16 + l15;
  const int n0     = tcPair * 32 + l15;

  // Per-thread 16B staging slot: 64 rows x 4 chunks of 8 bf16.
  const int sr = tid >> 2;              // row within tile
  const int sc = (tid & 3) << 3;        // bf16 col (0,8,16,24)
  const __bf16* gA0 = &A [(size_t)(mBase + sr) * KK_ + sc];
  const __bf16* gB0 = &BT[(size_t)(nBase + sr) * KK_ + sc];

  v8f c0 = {}; v8f c1 = {};

#if USE_ASYNC_LDS
  // Prologue: start DMA of tile 0 into buffer 0.
  __builtin_amdgcn_global_load_async_to_lds_b128(GPTR(gA0),
                                                 LPTR(&As[0][sr][sc]), 0, 0);
  __builtin_amdgcn_global_load_async_to_lds_b128(GPTR(gB0),
                                                 LPTR(&BsT[0][sr][sc]), 0, 0);
#endif

#pragma unroll
  for (int kb = 0; kb < KSTEPS_; ++kb) {
    const int cur = kb & 1;
#if USE_ASYNC_LDS
    if (kb + 1 < KSTEPS_) {
      // Kick off next tile's DMA into the other buffer, then drain only the
      // OLDER two copies (this tile) -- next tile's DMA overlaps the WMMAs.
      const int k1 = (kb + 1) << 5;
      __builtin_amdgcn_global_load_async_to_lds_b128(
          GPTR(gA0 + k1), LPTR(&As[cur ^ 1][sr][sc]), 0, 0);
      __builtin_amdgcn_global_load_async_to_lds_b128(
          GPTR(gB0 + k1), LPTR(&BsT[cur ^ 1][sr][sc]), 0, 0);
      __builtin_amdgcn_s_wait_asynccnt(2);
    } else {
      __builtin_amdgcn_s_wait_asynccnt(0);
    }
#else
    const int k0 = kb << 5;
    *(uint4*)&As [cur][sr][sc] = *(const uint4*)(gA0 + k0);
    *(uint4*)&BsT[cur][sr][sc] = *(const uint4*)(gB0 + k0);
    if (kb + 1 < KSTEPS_) {              // prefetch next tile
      __builtin_prefetch(gA0 + k0 + 32, 0, 0);
      __builtin_prefetch(gB0 + k0 + 32, 0, 0);
    }
#endif
    __syncthreads();   // this tile visible to all waves

    // ---- Fragments: two b128 LDS loads each. ----
    // A 16x32: elems 0-7 = K khalf..+7, elems 8-15 = K khalf+16..+23.
    FragU fa;
    fa.q[0] = *(const uint4*)&As[cur][mrow][khalf];
    fa.q[1] = *(const uint4*)&As[cur][mrow][khalf + 16];
    // B 32x16: elems 0-7 = K kboff..+7, elems 8-15 = K kboff+16..+23 at fixed N.
    FragU fb0, fb1;
    fb0.q[0] = *(const uint4*)&BsT[cur][n0][kboff];
    fb0.q[1] = *(const uint4*)&BsT[cur][n0][kboff + 16];
    fb1.q[0] = *(const uint4*)&BsT[cur][n0 + 16][kboff];
    fb1.q[1] = *(const uint4*)&BsT[cur][n0 + 16][kboff + 16];

    c0 = __builtin_amdgcn_wmma_f32_16x16x32_bf16(false, fa.v, false, fb0.v,
                                                 (short)0, c0, false, false);
    c1 = __builtin_amdgcn_wmma_f32_16x16x32_bf16(false, fa.v, false, fb1.v,
                                                 (short)0, c1, false, false);
    __syncthreads();   // all reads of this buffer done before it's re-filled
  }

  // C/D layout: N = lane&15 ; VGPR r -> M = r (+8 for lanes 16-31).
  const int mrow0 = mBase + tr * 16 + ((lane >> 4) << 3);
  const int ncol0 = nBase + tcPair * 32 + l15;
#pragma unroll
  for (int r = 0; r < 8; ++r) {
    float v0 = c0[r], v1 = c1[r];
    if (do_clip) {
      if (v0 != v0) v0 = 0.0f;
      if (v1 != v1) v1 = 0.0f;
      v0 = fminf(fmaxf(v0, -1000.0f), 1000.0f);
      v1 = fminf(fmaxf(v1, -1000.0f), 1000.0f);
    }
    C[(size_t)(mrow0 + r) * N + ncol0]      = v0;
    C[(size_t)(mrow0 + r) * N + ncol0 + 16] = v1;
  }
}

// ---------------------------------------------------------------------------
// Per-direction: causal depthwise conv (4 taps) + SiLU + x-projection (->48)
// + dt projection (16->256) + softplus.  One 256-thread block per (b, t).
// ---------------------------------------------------------------------------
__global__ __launch_bounds__(256) void conv_proj_kernel(
    const float* __restrict__ xz, const float* __restrict__ conv_w,
    const float* __restrict__ conv_b, const float* __restrict__ W_xproj,
    const float* __restrict__ W_dt, const float* __restrict__ b_dt,
    float* __restrict__ xc_buf, float* __restrict__ delta_buf,
    float* __restrict__ Bbuf, float* __restrict__ Cbuf, int dir) {
  __shared__ float xcs[DIN_];
  __shared__ float dbl[XPJ_];

  const int bt = blockIdx.x;
  const int b  = bt >> 12;        // / N_
  const int t  = bt & (N_ - 1);
  const int d  = threadIdx.x;

  // Causal depthwise conv along the direction's sequence order.
  float acc = 0.0f;
#pragma unroll
  for (int k = 0; k < DC_; ++k) {
    int ts = t - (DC_ - 1) + k;
    if (ts >= 0) {
      int tok = perm_tok(ts, dir);
      acc += xz[((size_t)b * N_ + tok) * (2 * DIN_) + d] * conv_w[d * DC_ + k];
    }
  }
  acc += conv_b[d];
  float xc = acc / (1.0f + __expf(-acc));   // SiLU
  xcs[d] = xc;
  xc_buf[((size_t)b * N_ + t) * DIN_ + d] = xc;
  __syncthreads();

  // dbl = xc @ W_xproj  (48 outputs of K=256)
  if (d < XPJ_) {
    float s = 0.0f;
#pragma unroll 4
    for (int r = 0; r < DIN_; ++r) s += xcs[r] * W_xproj[r * XPJ_ + d];
    dbl[d] = s;
    if (d >= DTR_ && d < DTR_ + DS_)
      Bbuf[((size_t)b * N_ + t) * DS_ + (d - DTR_)] = s;
    else if (d >= DTR_ + DS_)
      Cbuf[((size_t)b * N_ + t) * DS_ + (d - DTR_ - DS_)] = s;
  }
  __syncthreads();

  // delta = softplus(dt @ W_dt + b_dt)
  float dv = b_dt[d];
#pragma unroll
  for (int r = 0; r < DTR_; ++r) dv += dbl[r] * W_dt[r * DIN_ + d];
  float delta = (dv > 20.0f) ? dv : log1pf(__expf(dv));
  delta_buf[((size_t)b * N_ + t) * DIN_ + d] = delta;
}

// ---------------------------------------------------------------------------
// Selective scan.  Each half-wave (16 lanes) owns one (b, d) channel; lane s
// carries state s.  h = exp(delta*A)*h + delta*B*x ; y = sum_s h*C.
// y is gated by SiLU(z) + D skip and accumulated into yacc at the token.
// 1024 channels -> 512 waves -> 64 blocks of 256 threads.
// ---------------------------------------------------------------------------
__global__ __launch_bounds__(256) void scan_kernel(
    const float* __restrict__ xz, const float* __restrict__ xc_buf,
    const float* __restrict__ delta_buf, const float* __restrict__ Bbuf,
    const float* __restrict__ Cbuf, const float* __restrict__ A_log,
    const float* __restrict__ Dp, float* __restrict__ yacc, int dir) {
  const int lane = threadIdx.x & 31;
  const int wave = threadIdx.x >> 5;
  const int gw   = blockIdx.x * 8 + wave;
  const int pair = gw * 2 + (lane >> 4);     // 0..B_*DIN_-1
  const int b    = pair >> 8;                // / DIN_
  const int d    = pair & (DIN_ - 1);
  const int s    = lane & 15;

  const float Aval = -__expf(A_log[d * DS_ + s]);
  const float Dd   = Dp[d];
  float h = 0.0f;

  for (int t = 0; t < N_; ++t) {
    const size_t base = (size_t)b * N_ + t;
    const float delta = delta_buf[base * DIN_ + d];
    const float xcv   = xc_buf[base * DIN_ + d];
    const float Bv    = Bbuf[base * DS_ + s];
    const float Cv    = Cbuf[base * DS_ + s];
    const float dA    = __expf(delta * Aval);
    h = dA * h + delta * Bv * xcv;
    float y = h * Cv;
    // Reduce the 16 states within the half-wave (butterfly).
    y += __shfl_xor(y, 1, 16);
    y += __shfl_xor(y, 2, 16);
    y += __shfl_xor(y, 4, 16);
    y += __shfl_xor(y, 8, 16);
    if (s == 0) {
      const int tok = perm_tok(t, dir);
      const float zv = xz[((size_t)b * N_ + tok) * (2 * DIN_) + DIN_ + d];
      const float g  = zv / (1.0f + __expf(-zv));   // SiLU gate
      yacc[((size_t)b * N_ + tok) * DIN_ + d] += (y + Dd * xcv) * g;
    }
  }
}

// ---------------------------------------------------------------------------
// Host-side orchestration.
// ---------------------------------------------------------------------------
extern "C" void kernel_launch(void* const* d_in, const int* in_sizes, int n_in,
                              void* d_out, int out_size, void* d_ws,
                              size_t ws_size, hipStream_t stream) {
  (void)in_sizes; (void)n_in; (void)out_size; (void)ws_size;
  const float* x       = (const float*)d_in[0];
  const float* W_in    = (const float*)d_in[1];
  const float* conv_w  = (const float*)d_in[2];
  const float* conv_b  = (const float*)d_in[3];
  const float* W_xproj = (const float*)d_in[4];
  const float* W_dt    = (const float*)d_in[5];
  const float* b_dt    = (const float*)d_in[6];
  const float* A_log   = (const float*)d_in[7];
  const float* Dp      = (const float*)d_in[8];
  const float* W_out   = (const float*)d_in[9];

  const int M = B_ * N_;   // 16384

  float* ws = (float*)d_ws;
  size_t off = 0;
  float* xz        = ws + off; off += (size_t)B_ * N_ * 2 * DIN_;  // 8M floats
  float* xc_buf    = ws + off; off += (size_t)B_ * N_ * DIN_;
  float* delta_buf = ws + off; off += (size_t)B_ * N_ * DIN_;
  float* Bbuf      = ws + off; off += (size_t)B_ * N_ * DS_;
  float* Cbuf      = ws + off; off += (size_t)B_ * N_ * DS_;
  float* yacc      = ws + off; off += (size_t)B_ * N_ * DIN_;
  // bf16 region (each float slot = two bf16)
  __bf16* bws = (__bf16*)(ws + off);
  size_t boff = 0;
  __bf16* x_bf     = bws + boff; boff += (size_t)M * KK_;
  __bf16* yacc_bf  = bws + boff; boff += (size_t)M * KK_;
  __bf16* WinT_bf  = bws + boff; boff += (size_t)(2 * DIN_) * KK_;
  __bf16* WoutT_bf = bws + boff; boff += (size_t)DM_ * KK_;

  // --- Precondition operands: bf16 activations, transposed bf16 weights. ---
  cvt_bf16_scale<<<(M * KK_ / 4 + 255) / 256, 256, 0, stream>>>(
      x, x_bf, 1.0f, M * KK_ / 4);
  transpose_cvt_bf16<<<(KK_ * 2 * DIN_ + 255) / 256, 256, 0, stream>>>(
      W_in, WinT_bf, KK_, 2 * DIN_);
  transpose_cvt_bf16<<<(KK_ * DM_ + 255) / 256, 256, 0, stream>>>(
      W_out, WoutT_bf, KK_, DM_);

  // xz = x @ W_in  (direction-invariant; compute once).
  dim3 g1(2 * DIN_ / 64, M / 64);
  gemm_bf16_wmma<<<g1, 256, 0, stream>>>(x_bf, WinT_bf, xz, 2 * DIN_, 0);

  (void)hipMemsetAsync(yacc, 0, sizeof(float) * (size_t)B_ * N_ * DIN_,
                       stream);

  for (int dir = 0; dir < 4; ++dir) {
    conv_proj_kernel<<<B_ * N_, 256, 0, stream>>>(
        xz, conv_w, conv_b, W_xproj, W_dt, b_dt,
        xc_buf, delta_buf, Bbuf, Cbuf, dir);
    scan_kernel<<<(B_ * DIN_) / 2 / 8, 256, 0, stream>>>(
        xz, xc_buf, delta_buf, Bbuf, Cbuf, A_log, Dp, yacc, dir);
  }

  // out = clip(nan_to_num((yacc / 4) @ W_out))  — linearity folds the average.
  cvt_bf16_scale<<<(M * KK_ / 4 + 255) / 256, 256, 0, stream>>>(
      yacc, yacc_bf, 0.25f, M * KK_ / 4);
  dim3 g2(DM_ / 64, M / 64);
  gemm_bf16_wmma<<<g2, 256, 0, stream>>>(yacc_bf, WoutT_bf, (float*)d_out,
                                         DM_, 1);
}
